// ResidualVectorQuantize_4157528343152
// MI455X (gfx1250) — matrix operand order repaired and checked by reading, not compiled, not run on hardware
//
#include <hip/hip_runtime.h>
#include <hip/hip_bf16.h>

typedef __attribute__((ext_vector_type(16))) __bf16 v16bf;
typedef __attribute__((ext_vector_type(8)))  float  v8f;
typedef __attribute__((ext_vector_type(4)))  unsigned int v4u;
typedef __attribute__((ext_vector_type(8)))  int v8i;
typedef __attribute__((ext_vector_type(4)))  int v4i;

#define B_    16
#define D_    256
#define T_    8192
#define NCB   7
#define KCB   1024
#define CD    8
#define TILE  32      // t-columns per block
#define DS    260     // Rt row stride in dwords (t-major, 16B aligned, bank offset 4)
#define NW    2       // waves per block (16 columns each)
#define BLK   (NW*32)

#define LOSS_OFF ((size_t)B_*D_*T_)
#define IDX_OFF  (LOSS_OFF + 2)

// workspace layout
#define WS_WIN_BYTES   ((size_t)NCB*CD*D_*4)   // f32 normalized w_in  [i][c][d]
#define WS_WOUT_BYTES  ((size_t)NCB*D_*CD*2)   // bf16 normalized w_out [i][d][c]

// ---------------- prep: weight-norm + codebook normalize ----------------
__global__ void rvq_prep(const float* __restrict__ in_v, const float* __restrict__ in_g,
                         const float* __restrict__ out_v, const float* __restrict__ out_g,
                         const float* __restrict__ cbs,
                         float* __restrict__ winN, __bf16* __restrict__ woutB,
                         __bf16* __restrict__ cbnB) {
  int r = blockIdx.x * blockDim.x + threadIdx.x;
  if (r < NCB*CD) {                                   // w_in rows: norm over D
    int i = r / CD, c = r % CD;
    const float* v = in_v + ((size_t)i*CD + c) * D_;
    float s = 0.f;
    for (int d = 0; d < D_; ++d) s += v[d]*v[d];
    float sc = in_g[i*CD + c] * rsqrtf(s);
    float* o = winN + ((size_t)i*CD + c) * D_;
    for (int d = 0; d < D_; ++d) o[d] = v[d] * sc;
    return;
  }
  r -= NCB*CD;
  if (r < NCB*D_) {                                   // w_out rows: norm over CD
    int i = r / D_, d = r % D_;
    const float* v = out_v + ((size_t)i*D_ + d) * CD;
    float s = 0.f;
    for (int c = 0; c < CD; ++c) s += v[c]*v[c];
    float sc = out_g[i*D_ + d] * rsqrtf(s);
    __bf16* o = woutB + ((size_t)i*D_ + d) * CD;
    for (int c = 0; c < CD; ++c) o[c] = (__bf16)(v[c] * sc);
    return;
  }
  r -= NCB*D_;
  if (r < NCB*KCB) {                                  // codebook rows: l2norm
    int i = r / KCB, k = r % KCB;
    const float* v = cbs + ((size_t)i*KCB + k) * CD;
    float s = 0.f;
    for (int c = 0; c < CD; ++c) s += v[c]*v[c];
    float n = fmaxf(sqrtf(s), 1e-12f);
    float sc = 1.f / n;
    __bf16* o = cbnB + ((size_t)i*KCB + k) * CD;
    for (int c = 0; c < CD; ++c) o[c] = (__bf16)(v[c] * sc);
  }
}

__global__ void rvq_init(float* out) { out[LOSS_OFF] = 0.f; out[LOSS_OFF+1] = 0.f; }

// ---------------- fused RVQ main kernel ----------------
__launch_bounds__(BLK)
__global__ void rvq_main(const float* __restrict__ z,
                         const float* __restrict__ in_b,
                         const float* __restrict__ out_b,
                         const float* __restrict__ cbs,
                         const int*   __restrict__ nq_p,
                         const float* __restrict__ winN,
                         const __bf16* __restrict__ woutB,
                         const __bf16* __restrict__ cbnB,
                         float* __restrict__ out) {
  __shared__ float  Rt[TILE*DS];      // residual tile fp32, t-major [t][d]
  __shared__ __bf16 CBnS[KCB*CD];     // normalized codebook (bf16) for current CB
  __shared__ float  ZeS[NW][16*CD];   // z_e scratch [t_local][c] per wave
  __shared__ int    IdxS[NW][16];

  const int tid  = threadIdx.x;
  const int lane = tid & 31, wv = tid >> 5;
  const int l16  = lane & 15, half = lane >> 4;
  const int nq   = *nq_p;
  const int nqc  = (nq < NCB) ? nq : NCB;

  const int blk = blockIdx.x;
  const int b   = blk / (T_/TILE);
  const int t0  = (blk % (T_/TILE)) * TILE;

  // load residual tile (z) into LDS, transposed to t-major
  for (int e = tid; e < D_*(TILE/4); e += BLK) {
    int d = e / (TILE/4), q = e % (TILE/4);
    float4 v = *(const float4*)(z + ((size_t)b*D_ + d)*T_ + t0 + q*4);
    Rt[(q*4+0)*DS + d] = v.x;
    Rt[(q*4+1)*DS + d] = v.y;
    Rt[(q*4+2)*DS + d] = v.z;
    Rt[(q*4+3)*DS + d] = v.w;
  }
  if (nq <= 0) {
    for (int e = tid; e < D_*TILE; e += BLK) {
      int d = e / TILE, q = e % TILE;
      out[((size_t)b*D_ + d)*T_ + t0 + q] = 0.f;
    }
  }
  __syncthreads();

  const float lscale = 1.0f / (float)((size_t)B_*CD*T_);
  const int twv = wv * 16;  // this wave's t-base within tile

  for (int i = 0; i < NCB; ++i) {
    // ---- TDM: DMA the 16KB normalized codebook into LDS (wave 0 only) ----
    if (wv == 0) {
      unsigned lds_off = (unsigned)(unsigned long long)&CBnS[0];
      unsigned long long ga = (unsigned long long)(cbnB + (size_t)i*KCB*CD);
      v4u g0; g0[0] = 1u;                       // count=1, user descriptor
      g0[1] = lds_off;                          // lds_addr
      g0[2] = (unsigned)ga;                     // global_addr[31:0]
      g0[3] = (unsigned)((ga >> 32) & 0x1FFFFFFull) | (2u << 30);  // addr[56:32] | type=2
      v8i g1;
      g1[0] = (3 << 16);                        // wg_mask=0, data_size=8B
      g1[1] = (int)(2048u << 16);               // tensor_dim0[15:0]=2048 (8B units)
      g1[2] = (1 << 16);                        // tensor_dim0 hi=0, tensor_dim1=1
      g1[3] = (int)(2048u << 16);               // tile_dim0=2048
      g1[4] = 0;                                // tile_dim1=0, tile_dim2=0
      g1[5] = 2048;                             // tensor_dim0_stride lo
      g1[6] = 0; g1[7] = 0;
      v4i gz4 = {0, 0, 0, 0};
      v8i gz8 = {0, 0, 0, 0, 0, 0, 0, 0};
      __builtin_amdgcn_tensor_load_to_lds(g0, g1, gz4, gz4, gz8, 0);
      __builtin_amdgcn_s_wait_tensorcnt(0);
    }
    __syncthreads();

    // ---- preload B blocks (w_in^T, from L2) for in_proj GEMM ----
    v16bf Bw[8];
    {
      int c = l16;
      const float* wbase = winN + ((size_t)i*CD + (c < CD ? c : 0)) * D_;
      #pragma unroll
      for (int k = 0; k < 8; ++k) {
        v16bf t = {};
        if (c < CD) {
          const float4* p = (const float4*)(wbase + k*32 + half*16);
          float4 a0 = p[0], a1 = p[1], a2 = p[2], a3 = p[3];
          t[0]=(__bf16)a0.x; t[1]=(__bf16)a0.y; t[2]=(__bf16)a0.z; t[3]=(__bf16)a0.w;
          t[4]=(__bf16)a1.x; t[5]=(__bf16)a1.y; t[6]=(__bf16)a1.z; t[7]=(__bf16)a1.w;
          t[8]=(__bf16)a2.x; t[9]=(__bf16)a2.y; t[10]=(__bf16)a2.z; t[11]=(__bf16)a2.w;
          t[12]=(__bf16)a3.x; t[13]=(__bf16)a3.y; t[14]=(__bf16)a3.z; t[15]=(__bf16)a3.w;
        }
        Bw[k] = t;
      }
    }

    // ---- z_e^T[t,c] = residual^T x w_in^T  (8x WMMA over K=256) ----
    v8f acc = {};
    {
      const float* rrow = &Rt[(twv + l16)*DS];
      #pragma unroll
      for (int k = 0; k < 8; ++k) {
        int kb = k*32 + half*16;
        const float4* p0 = (const float4*)(rrow + kb);
        const float4* p1 = (const float4*)(rrow + kb + 8);
        float4 x0 = p0[0], x1 = p0[1];
        float4 y0 = p1[0], y1 = p1[1];
        v16bf a;
        a[0]=(__bf16)x0.x; a[1]=(__bf16)x0.y; a[2]=(__bf16)x0.z; a[3]=(__bf16)x0.w;
        a[4]=(__bf16)x1.x; a[5]=(__bf16)x1.y; a[6]=(__bf16)x1.z; a[7]=(__bf16)x1.w;
        a[8]=(__bf16)y0.x; a[9]=(__bf16)y0.y; a[10]=(__bf16)y0.z; a[11]=(__bf16)y0.w;
        a[12]=(__bf16)y1.x; a[13]=(__bf16)y1.y; a[14]=(__bf16)y1.z; a[15]=(__bf16)y1.w;
        acc = __builtin_amdgcn_wmma_f32_16x16x32_bf16(false, a, false, Bw[k],
                                                      (short)0, acc, false, false);
      }
    }
    {
      float bias = (l16 < CD) ? in_b[i*CD + l16] : 0.f;
      #pragma unroll
      for (int r = 0; r < 8; ++r) acc[r] += bias;
    }

    // ---- scatter z_e to LDS scratch [t_local][c] ----
    if (l16 < CD) {
      #pragma unroll
      for (int r = 0; r < 8; ++r)
        ZeS[wv][(r + half*8)*CD + l16] = acc[r];
    }
    __syncthreads();

    // ---- build A_enc (K = cd, zero-padded to 32) ----
    v16bf Aenc = {};
    if (half == 0) {
      const float* p = &ZeS[wv][l16*CD];
      #pragma unroll
      for (int e = 0; e < CD; ++e) Aenc[e] = (__bf16)p[e];
    }

    // ---- distance scores vs all 1024 codes (LDS); running per-slot argmax ----
    v8f bestV;
    int bestI[8];
    #pragma unroll
    for (int r = 0; r < 8; ++r) { bestV[r] = -3.0e38f; bestI[r] = 0; }
    #pragma clang loop unroll_count(4)
    for (int j = 0; j < KCB/16; ++j) {
      v16bf Bc = {};
      if (half == 0) {
        uint4 raw = *(const uint4*)&CBnS[(size_t)(j*16 + l16)*CD];
        const __bf16* q = (const __bf16*)&raw;
        #pragma unroll
        for (int e = 0; e < CD; ++e) Bc[e] = q[e];
      }
      v8f zc = {};
      v8f sc = __builtin_amdgcn_wmma_f32_16x16x32_bf16(false, Aenc, false, Bc,
                                                       (short)0, zc, false, false);
      int cand = j*16 + l16;
      #pragma unroll
      for (int r = 0; r < 8; ++r) {
        bool gt = sc[r] > bestV[r];
        bestV[r] = gt ? sc[r] : bestV[r];
        bestI[r] = gt ? cand  : bestI[r];
      }
    }

    // ---- cross-lane argmax reduce within each 16-lane half ----
    #pragma unroll
    for (int m = 1; m < 16; m <<= 1) {
      #pragma unroll
      for (int r = 0; r < 8; ++r) {
        float ov = __shfl_xor(bestV[r], m, 32);
        int   oi = __shfl_xor(bestI[r], m, 32);
        bool take = (ov > bestV[r]) || (ov == bestV[r] && oi < bestI[r]);
        bestV[r] = take ? ov : bestV[r];
        bestI[r] = take ? oi : bestI[r];
      }
    }
    if (l16 == 0) {
      #pragma unroll
      for (int r = 0; r < 8; ++r) {
        int tl = r + half*8;
        IdxS[wv][tl] = bestI[r];
        out[IDX_OFF + ((size_t)b*NCB + i)*T_ + t0 + twv + tl] = (float)bestI[r];
      }
    }
    __syncthreads();

    // ---- gather raw code rows -> A_q; loss partial ----
    v16bf Aq = {};
    float lsum = 0.f;
    if (half == 0) {
      int idx = IdxS[wv][l16];
      const float* cb = cbs + ((size_t)i*KCB + idx)*CD;
      const float* ze = &ZeS[wv][l16*CD];
      #pragma unroll
      for (int e = 0; e < CD; ++e) {
        float cv = cb[e];
        Aq[e] = (__bf16)cv;
        float dlt = ze[e] - cv;
        lsum += dlt*dlt;
      }
    }
    #pragma unroll
    for (int m = 1; m < 32; m <<= 1) lsum += __shfl_xor(lsum, m, 32);
    if (lane == 0 && i < nq) {
      float v = lsum * lscale;          // commit and codebook losses identical fwd
      unsafeAtomicAdd(&out[LOSS_OFF],   v);
      unsafeAtomicAdd(&out[LOSS_OFF+1], v);
    }

    // ---- out_proj (16x WMMA over d, weights from L2) + residual update ----
    {
      const __bf16* wo_i = woutB + (size_t)i*D_*CD;
      const float*  ob_i = out_b + (size_t)i*D_;
      for (int dj = 0; dj < D_/16; ++dj) {
        v16bf Bo = {};
        if (half == 0) {
          uint4 raw = *(const uint4*)(wo_i + (size_t)(dj*16 + l16)*CD);
          const __bf16* q = (const __bf16*)&raw;
          #pragma unroll
          for (int e = 0; e < CD; ++e) Bo[e] = q[e];
        }
        v8f zc = {};
        v8f o = __builtin_amdgcn_wmma_f32_16x16x32_bf16(false, Aq, false, Bo,
                                                        (short)0, zc, false, false);
        int d = dj*16 + l16;
        float ob = ob_i[d];
        #pragma unroll
        for (int r = 0; r < 8; ++r) {
          int t = twv + r + half*8;     // unique (t,d) per lane -> plain RMW
          Rt[t*DS + d] -= (o[r] + ob);
        }
      }
    }
    __syncthreads();

    // ---- emit z_q = z - residual when reaching n_quantizers ----
    if (i + 1 == nqc) {
      for (int e = tid; e < D_*(TILE/4); e += BLK) {
        int d = e / (TILE/4), q = e % (TILE/4);
        float4 v = *(const float4*)(z + ((size_t)b*D_ + d)*T_ + t0 + q*4);
        float4 w;
        w.x = v.x - Rt[(q*4+0)*DS + d];
        w.y = v.y - Rt[(q*4+1)*DS + d];
        w.z = v.z - Rt[(q*4+2)*DS + d];
        w.w = v.w - Rt[(q*4+3)*DS + d];
        *(float4*)(out + ((size_t)b*D_ + d)*T_ + t0 + q*4) = w;
      }
      __syncthreads();
    }
  }
}

extern "C" void kernel_launch(void* const* d_in, const int* in_sizes, int n_in,
                              void* d_out, int out_size, void* d_ws, size_t ws_size,
                              hipStream_t stream) {
  const float* z     = (const float*)d_in[0];
  const float* in_v  = (const float*)d_in[1];
  const float* in_g  = (const float*)d_in[2];
  const float* in_b  = (const float*)d_in[3];
  const float* out_v = (const float*)d_in[4];
  const float* out_g = (const float*)d_in[5];
  const float* out_b = (const float*)d_in[6];
  const float* cbs   = (const float*)d_in[7];
  const int*   nq    = (const int*)d_in[8];

  float*  winN  = (float*)d_ws;
  __bf16* woutB = (__bf16*)((char*)d_ws + WS_WIN_BYTES);
  __bf16* cbnB  = (__bf16*)((char*)d_ws + WS_WIN_BYTES + WS_WOUT_BYTES);
  float*  out   = (float*)d_out;

  rvq_init<<<1, 1, 0, stream>>>(out);
  int rows = NCB*CD + NCB*D_ + NCB*KCB;
  rvq_prep<<<(rows + 255)/256, 256, 0, stream>>>(in_v, in_g, out_v, out_g, cbs,
                                                 winN, woutB, cbnB);
  rvq_main<<<B_*(T_/TILE), BLK, 0, stream>>>(z, in_b, out_b, cbs, nq,
                                             winN, woutB, cbnB, out);
}
